// PoseGuidedMultipoleFusion_15659450761395
// MI455X (gfx1250) — compile-verified, matrix operand order checked by reading, not run on hardware
//
#include <hip/hip_runtime.h>
#include <hip/hip_bf16.h>

// ---------------------------------------------------------------------------
// PoseGuidedMultipoleFusion for MI455X (gfx1250, wave32, WMMA).
// All matmuls (conv-as-GEMM, Q/K/V/O projections, FFN, attention S & AV) go
// through v_wmma_f32_16x16x32_f16 with fp32 accumulation.
// ---------------------------------------------------------------------------

typedef __attribute__((ext_vector_type(16))) _Float16 v16h;
typedef __attribute__((ext_vector_type(8)))  _Float16 v8h;
typedef __attribute__((ext_vector_type(8)))  float    v8f;
typedef __attribute__((ext_vector_type(4)))  float    v4f;

// Load a 16x32 f16 A/B fragment row chunk from LDS.
// Per CDNA5 ISA layout: lane half 0 holds K {kb..kb+7} and {kb+16..kb+23},
// half 1 holds K {kb+8..kb+15} and {kb+24..kb+31}; each chunk is 16B.
__device__ __forceinline__ v16h lds_frag(const _Float16* rowp, int kbase, int hf) {
  v8h c0 = *(const v8h*)(rowp + kbase + hf * 8);
  v8h c1 = *(const v8h*)(rowp + kbase + 16 + hf * 8);
  v16h r;
#pragma unroll
  for (int i = 0; i < 8; ++i) { r[i] = c0[i]; r[i + 8] = c1[i]; }
  return r;
}

__device__ __forceinline__ v8f wmma_f16(v16h a, v16h b, v8f c) {
  return __builtin_amdgcn_wmma_f32_16x16x32_f16(
      /*neg_a=*/false, a, /*neg_b=*/false, b,
      /*c_mod=*/(short)0, c, /*reuse_a=*/false, /*reuse_b=*/false);
}

// ---------------------------------------------------------------------------
// Generic GEMM:  C[M,N] = op( A[M,K] * B + bias[N] ),  f16 WMMA, fp32 accum.
//   BT:   B stored row-major [N,K] (else [K,N])
//   RELU: apply relu in epilogue
//   ACC:  C += alpha * (A*B + bias), alpha read from device ptr (nullptr -> 1)
// Block tile 128x64, BK=32, 256 threads = 8 waves, 2x2 WMMA tiles per wave.
// ---------------------------------------------------------------------------
template <bool BT, bool RELU, bool ACC>
__global__ __launch_bounds__(256) void gemm_wmma_f32(
    const float* __restrict__ A, const float* __restrict__ Bm,
    const float* __restrict__ bias, const float* __restrict__ alphap,
    float* __restrict__ C, int M, int N, int K) {
  constexpr int BM = 128, BN = 64, BK = 32, LDA = 40, LDB = 40;  // padded strides
  __shared__ _Float16 As[BM * LDA];
  __shared__ _Float16 Bs[BN * LDB];

  const int tid = threadIdx.x;
  const int bM = blockIdx.y * BM, bN = blockIdx.x * BN;
  const int w = tid >> 5, lane = tid & 31, hf = lane >> 4, lm = lane & 15;
  const int wm = w & 3, wn = w >> 2;

  float alpha = 1.0f;
  if (ACC && alphap) alpha = *alphap;

  v8f acc[2][2] = {};

  const int arow = tid >> 1;          // 0..127
  const int akseg = (tid & 1) * 16;   // 0 / 16

  for (int kk = 0; kk < K; kk += BK) {
    // ---- stage A tile (fp32 -> f16) ----
    {
      const int gm = bM + arow;
      v4f a0 = {}, a1 = {}, a2 = {}, a3 = {};
      if (gm < M) {
        const float* p = A + (size_t)gm * K + kk + akseg;
        a0 = *(const v4f*)(p);      a1 = *(const v4f*)(p + 4);
        a2 = *(const v4f*)(p + 8);  a3 = *(const v4f*)(p + 12);
      }
      v8h h0, h1;
#pragma unroll
      for (int i = 0; i < 4; ++i) {
        h0[i] = (_Float16)a0[i]; h0[i + 4] = (_Float16)a1[i];
        h1[i] = (_Float16)a2[i]; h1[i + 4] = (_Float16)a3[i];
      }
      _Float16* d = As + arow * LDA + akseg;
      *(v8h*)(d) = h0;
      *(v8h*)(d + 8) = h1;
    }
    // ---- stage B tile into [N][K] f16 layout ----
    if (BT) {
      const int bn = tid >> 2, kseg = (tid & 3) * 8;
      const float* p = Bm + (size_t)(bN + bn) * K + kk + kseg;
      v4f b0 = *(const v4f*)p, b1 = *(const v4f*)(p + 4);
      v8h h;
#pragma unroll
      for (int i = 0; i < 4; ++i) { h[i] = (_Float16)b0[i]; h[i + 4] = (_Float16)b1[i]; }
      *(v8h*)(Bs + bn * LDB + kseg) = h;
    } else {
      const int bk = tid >> 3, nseg = (tid & 7) * 8;
      const float* p = Bm + (size_t)(kk + bk) * N + bN + nseg;
      v4f b0 = *(const v4f*)p, b1 = *(const v4f*)(p + 4);
#pragma unroll
      for (int j = 0; j < 4; ++j) Bs[(nseg + j) * LDB + bk] = (_Float16)b0[j];
#pragma unroll
      for (int j = 0; j < 4; ++j) Bs[(nseg + 4 + j) * LDB + bk] = (_Float16)b1[j];
    }
    __syncthreads();

    // ---- WMMA compute ----
    v16h af[2], bf[2];
#pragma unroll
    for (int mt = 0; mt < 2; ++mt)
      af[mt] = lds_frag(As + (wm * 32 + mt * 16 + lm) * LDA, 0, hf);
#pragma unroll
    for (int nt = 0; nt < 2; ++nt)
      bf[nt] = lds_frag(Bs + (wn * 32 + nt * 16 + lm) * LDB, 0, hf);
#pragma unroll
    for (int mt = 0; mt < 2; ++mt)
#pragma unroll
      for (int nt = 0; nt < 2; ++nt)
        acc[mt][nt] = wmma_f16(af[mt], bf[nt], acc[mt][nt]);
    __syncthreads();
  }

  // ---- epilogue (C/D layout: lane n = lm + 16*?, row m = r + 8*hf) ----
#pragma unroll
  for (int mt = 0; mt < 2; ++mt)
#pragma unroll
    for (int nt = 0; nt < 2; ++nt) {
      const int gn = bN + wn * 32 + nt * 16 + lm;
      const float bv = bias ? bias[gn] : 0.0f;
#pragma unroll
      for (int r = 0; r < 8; ++r) {
        const int gm = bM + wm * 32 + mt * 16 + hf * 8 + r;
        if (gm < M) {
          float v = acc[mt][nt][r] + bv;
          if (RELU) v = fmaxf(v, 0.0f);
          if (ACC) C[(size_t)gm * N + gn] += alpha * v;
          else     C[(size_t)gm * N + gn] = v;
        }
      }
    }
}

// ---------------------------------------------------------------------------
// Flash-style attention per (b, head).  q:[100,96] padded to 112, keys tiled
// by 64.  S = QK^T via WMMA; online softmax (row reductions stay inside the
// 16-lane half that owns the row: shfl_xor masks 8..1); O += P*V via WMMA.
// ---------------------------------------------------------------------------
__global__ __launch_bounds__(256) void attn_kernel(
    const float* __restrict__ qbuf, const float* __restrict__ kbuf,
    const float* __restrict__ vbuf, const float* __restrict__ biasb,
    float* __restrict__ outb, int HW, float scale) {
  constexpr int NQ = 100, NQP = 112, DH = 96, TK = 64;
  constexpr int LQ = 104, LK = 104, LV = 72, LP = 72;  // padded f16 strides
  __shared__ _Float16 qs[NQP * LQ];
  __shared__ _Float16 kps[8064];       // aliased: K tile [64*LK] then P [7*16*LP]
  __shared__ _Float16 vts[DH * LV];    // V^T: [d][key]
  __shared__ float bs[TK];

  const int tid = threadIdx.x;
  const int b = blockIdx.x >> 3, h = blockIdx.x & 7;
  const int w = tid >> 5, lane = tid & 31, hf = lane >> 4, lm = lane & 15;

  // stage Q (zero-pad rows 100..111)
  for (int idx = tid; idx < NQP * DH; idx += 256) {
    const int qr = idx / DH, d = idx % DH;
    float v = (qr < NQ) ? qbuf[((size_t)b * NQ + qr) * 768 + h * DH + d] : 0.0f;
    qs[qr * LQ + d] = (_Float16)v;
  }

  float mstate[8], lstate[8];
#pragma unroll
  for (int r = 0; r < 8; ++r) { mstate[r] = -1e30f; lstate[r] = 0.0f; }
  v8f oacc[6] = {};

  __syncthreads();

  for (int kt = 0; kt < HW; kt += TK) {
    // stage K tile ([n][k]) and V^T tile ([d][n])
    for (int idx = tid; idx < TK * DH; idx += 256) {
      const int n = idx / DH, d = idx % DH;
      const size_t row = ((size_t)b * HW + kt + n) * 768 + h * DH + d;
      kps[n * LK + d] = (_Float16)kbuf[row];
      vts[d * LV + n] = (_Float16)vbuf[row];
    }
    if (tid < TK) bs[tid] = biasb[(size_t)b * HW + kt + tid];
    __syncthreads();

    float pv[4][8];
    if (w < 7) {
      v8f sacc[4] = {};
#pragma unroll
      for (int ks = 0; ks < 3; ++ks) {
        const v16h aq = lds_frag(qs + (w * 16 + lm) * LQ, ks * 32, hf);
#pragma unroll
        for (int nt = 0; nt < 4; ++nt) {
          const v16h bk2 = lds_frag(kps + (nt * 16 + lm) * LK, ks * 32, hf);
          sacc[nt] = wmma_f16(aq, bk2, sacc[nt]);
        }
      }
#pragma unroll
      for (int r = 0; r < 8; ++r) {
        float rowmax = -1e30f;
#pragma unroll
        for (int nt = 0; nt < 4; ++nt) {
          const float s = sacc[nt][r] * scale + bs[nt * 16 + lm];
          pv[nt][r] = s;
          rowmax = fmaxf(rowmax, s);
        }
#pragma unroll
        for (int m2 = 8; m2 >= 1; m2 >>= 1)
          rowmax = fmaxf(rowmax, __shfl_xor(rowmax, m2, 32));
        const float mnew = fmaxf(mstate[r], rowmax);
        float rowsum = 0.0f;
#pragma unroll
        for (int nt = 0; nt < 4; ++nt) {
          const float p = __expf(pv[nt][r] - mnew);
          pv[nt][r] = p;
          rowsum += p;
        }
#pragma unroll
        for (int m2 = 8; m2 >= 1; m2 >>= 1)
          rowsum += __shfl_xor(rowsum, m2, 32);
        const float fac = __expf(mstate[r] - mnew);
        lstate[r] = lstate[r] * fac + rowsum;
        mstate[r] = mnew;
#pragma unroll
        for (int n2 = 0; n2 < 6; ++n2) oacc[n2][r] *= fac;
      }
    }
    __syncthreads();  // K-tile reads done; kps may be reused as P scratch

    if (w < 7) {
#pragma unroll
      for (int nt = 0; nt < 4; ++nt)
#pragma unroll
        for (int r = 0; r < 8; ++r)
          kps[(w * 16 + hf * 8 + r) * LP + nt * 16 + lm] = (_Float16)pv[nt][r];
    }
    __syncthreads();  // P visible

    if (w < 7) {
#pragma unroll
      for (int kp = 0; kp < 2; ++kp) {
        const v16h pa = lds_frag(kps + (w * 16 + lm) * LP, kp * 32, hf);
#pragma unroll
        for (int n2 = 0; n2 < 6; ++n2) {
          const v16h vbf = lds_frag(vts + (n2 * 16 + lm) * LV, kp * 32, hf);
          oacc[n2] = wmma_f16(pa, vbf, oacc[n2]);
        }
      }
    }
    __syncthreads();  // done with vts / P before next staging
  }

  if (w < 7) {
#pragma unroll
    for (int n2 = 0; n2 < 6; ++n2)
#pragma unroll
      for (int r = 0; r < 8; ++r) {
        const int qr = w * 16 + hf * 8 + r;
        if (qr < NQ)
          outb[((size_t)b * NQ + qr) * 768 + h * DH + n2 * 16 + lm] =
              oacc[n2][r] / lstate[r];
      }
  }
}

// ---------------------------------------------------------------------------
// LayerNorm over C=768 (3 elems/thread, 256 threads/row), optional residual.
// ---------------------------------------------------------------------------
__global__ __launch_bounds__(256) void ln_kernel(
    const float* __restrict__ x, const float* __restrict__ res,
    const float* __restrict__ g, const float* __restrict__ be,
    float* __restrict__ y) {
  const size_t row = blockIdx.x;
  const int tid = threadIdx.x;
  float v[3];
#pragma unroll
  for (int j = 0; j < 3; ++j) {
    const int c = tid + j * 256;
    float t = x[row * 768 + c];
    if (res) t += res[row * 768 + c];
    v[j] = t;
  }
  float s = v[0] + v[1] + v[2];
#pragma unroll
  for (int m = 16; m >= 1; m >>= 1) s += __shfl_xor(s, m, 32);
  __shared__ float sb[8], sb2[8], mv[2];
  if ((tid & 31) == 0) sb[tid >> 5] = s;
  __syncthreads();
  if (tid == 0) {
    float t = 0.0f;
    for (int i = 0; i < 8; ++i) t += sb[i];
    mv[0] = t * (1.0f / 768.0f);
  }
  __syncthreads();
  const float mean = mv[0];
  float q = 0.0f;
#pragma unroll
  for (int j = 0; j < 3; ++j) { const float d = v[j] - mean; q += d * d; }
#pragma unroll
  for (int m = 16; m >= 1; m >>= 1) q += __shfl_xor(q, m, 32);
  if ((tid & 31) == 0) sb2[tid >> 5] = q;
  __syncthreads();
  if (tid == 0) {
    float t = 0.0f;
    for (int i = 0; i < 8; ++i) t += sb2[i];
    mv[1] = t * (1.0f / 768.0f);
  }
  __syncthreads();
  const float rstd = rsqrtf(mv[1] + 1e-5f);
#pragma unroll
  for (int j = 0; j < 3; ++j) {
    const int c = tid + j * 256;
    y[row * 768 + c] = (v[j] - mean) * rstd * g[c] + be[c];
  }
}

// keypoint distance bias: out[b,p] = max(-mind/(2*sigma^2), -10000)
__global__ __launch_bounds__(256) void bias_k(
    const float* __restrict__ kp, float* __restrict__ o, int H, int W,
    float inv2s2, int HW) {
  const int idx = blockIdx.x * 256 + threadIdx.x;
  if (idx >= 8 * HW) return;
  const int b = idx / HW, p = idx % HW;
  const int yy = p / W, xx = p % W;
  const float gx = (float)xx / (float)(W - 1), gy = (float)yy / (float)(H - 1);
  float m = 1e30f;
#pragma unroll
  for (int i = 0; i < 17; ++i) {
    const float kx = fminf(fmaxf(kp[(b * 17 + i) * 2 + 0], 0.0f), 1.0f);
    const float ky = fminf(fmaxf(kp[(b * 17 + i) * 2 + 1], 0.0f), 1.0f);
    const float dx = kx - gx, dy = ky - gy;
    m = fminf(m, dx * dx + dy * dy);
  }
  o[idx] = fmaxf(-m * inv2s2, -10000.0f);
}

// NCHW -> [B][HW][C] tiled transpose
__global__ void transpose_bc(const float* __restrict__ in,
                             float* __restrict__ out, int P) {
  __shared__ float t[32][33];
  const int b = blockIdx.z;
  const int p0 = blockIdx.x * 32, c0 = blockIdx.y * 32;
  for (int j = threadIdx.y; j < 32; j += 8)
    t[j][threadIdx.x] = in[((size_t)b * 768 + c0 + j) * P + p0 + threadIdx.x];
  __syncthreads();
  for (int j = threadIdx.y; j < 32; j += 8)
    out[((size_t)b * P + p0 + j) * 768 + c0 + threadIdx.x] = t[threadIdx.x][j];
}

// im2col for 2x2/stride-2 conv; K order = i*4 + ky*2 + kx (matches OIHW flat).
// mode 0: src NCHW; mode 1: src channels-last [B][Hin*Win][C]
__global__ __launch_bounds__(256) void im2col_k(
    const float* __restrict__ src, float* __restrict__ dst, int Hin, int Ho,
    int mode, size_t total) {
  const size_t idx = (size_t)blockIdx.x * 256 + threadIdx.x;
  if (idx >= total) return;
  const int c = (int)(idx % 3072);
  const size_t r = idx / 3072;
  const int xx = (int)(r % Ho);
  const int yy = (int)((r / Ho) % Ho);
  const int b = (int)(r / ((size_t)Ho * Ho));
  const int i = c >> 2, ky = (c >> 1) & 1, kx = c & 1;
  const int sy = 2 * yy + ky, sx = 2 * xx + kx;
  float v;
  if (mode == 0)
    v = src[(((size_t)b * 768 + i) * Hin + sy) * Hin + sx];
  else
    v = src[((size_t)b * Hin * Hin + (size_t)sy * Hin + sx) * 768 + i];
  dst[idx] = v;
}

__global__ void softmax3_k(const float* __restrict__ sw, float* __restrict__ o) {
  if (threadIdx.x == 0) {
    const float a = sw[0], b = sw[1], c = sw[2];
    const float m = fmaxf(a, fmaxf(b, c));
    const float ea = __expf(a - m), eb = __expf(b - m), ec = __expf(c - m);
    const float s = ea + eb + ec;
    o[0] = ea / s; o[1] = eb / s; o[2] = ec / s;
  }
}

__global__ __launch_bounds__(256) void zero_k(float* __restrict__ p, int n) {
  const int i = blockIdx.x * 256 + threadIdx.x;
  if (i < n) p[i] = 0.0f;
}

// ---------------------------------------------------------------------------
extern "C" void kernel_launch(void* const* d_in, const int* in_sizes, int n_in,
                              void* d_out, int out_size, void* d_ws,
                              size_t ws_size, hipStream_t stream) {
  const float* qtok = (const float*)d_in[0];
  const float* img  = (const float*)d_in[1];
  const float* kpts = (const float*)d_in[2];
  const float* nfg  = (const float*)d_in[3];
  const float* nfb  = (const float*)d_in[4];
  const float* dsw  = (const float*)d_in[5];
  const float* dsb  = (const float*)d_in[6];
  const float* qw   = (const float*)d_in[7];
  const float* qb   = (const float*)d_in[8];
  const float* kw   = (const float*)d_in[9];
  const float* kb   = (const float*)d_in[10];
  const float* vw   = (const float*)d_in[11];
  const float* vb   = (const float*)d_in[12];
  const float* ow   = (const float*)d_in[13];
  const float* ob   = (const float*)d_in[14];
  const float* sw   = (const float*)d_in[15];
  const float* nag  = (const float*)d_in[16];
  const float* nab  = (const float*)d_in[17];
  const float* fw1  = (const float*)d_in[18];
  const float* fb1  = (const float*)d_in[19];
  const float* fw2  = (const float*)d_in[20];
  const float* fb2  = (const float*)d_in[21];
  const float* ng2  = (const float*)d_in[22];
  const float* nb2  = (const float*)d_in[23];
  float* ws = (float*)d_ws;
  float* out = (float*)d_out;

  // workspace layout (floats); peak ~89M floats (~356 MB)
  size_t off = 0;
  auto alloc = [&](size_t n) { size_t o = off; off += (n + 255) & ~255ull; return o; };
  const size_t o_cl1  = alloc(8ull * 1024 * 768);
  const size_t o_cl2  = alloc(8ull * 256 * 768);
  const size_t o_feat = alloc(8ull * 4096 * 768);
  const size_t o_kbuf = alloc(8ull * 4096 * 768);  // also im2col / CL-transpose temp
  const size_t o_vbuf = alloc(8ull * 4096 * 768);
  const size_t o_qbuf = alloc(800ull * 768);
  const size_t o_attn = alloc(800ull * 768);
  const size_t o_fus  = alloc(800ull * 768);
  const size_t o_x    = alloc(800ull * 768);
  const size_t o_y    = alloc(800ull * 768);
  const size_t o_hid  = alloc(800ull * 3072);
  const size_t o_bias = alloc(8ull * 4096);
  const size_t o_wsm  = alloc(64);
  (void)in_sizes; (void)n_in; (void)out_size; (void)ws_size;

  softmax3_k<<<1, 32, 0, stream>>>(sw, ws + o_wsm);
  zero_k<<<(800 * 768 + 255) / 256, 256, 0, stream>>>(ws + o_fus, 800 * 768);

  const float sigmas[3] = {0.05f, 0.15f, 0.3f};
  const float qscale = 0.10206207261596577f;  // 96^-0.5

  for (int s = 0; s < 3; ++s) {
    const int Hc = 64 >> s, HW = Hc * Hc, M = 8 * HW;
    const float* clp;
    if (s == 0) {
      transpose_bc<<<dim3(HW / 32, 24, 8), dim3(32, 8), 0, stream>>>(
          img, ws + o_kbuf, HW);
      clp = ws + o_kbuf;
    } else {
      const float* src = (s == 1) ? img : (ws + o_cl1);
      const int Hin = Hc * 2;
      const size_t tot = (size_t)M * 3072;
      im2col_k<<<(unsigned)((tot + 255) / 256), 256, 0, stream>>>(
          src, ws + o_kbuf, Hin, Hc, (s == 1) ? 0 : 1, tot);
      float* cls = (s == 1) ? (ws + o_cl1) : (ws + o_cl2);
      gemm_wmma_f32<true, false, false><<<dim3(12, M / 128), 256, 0, stream>>>(
          ws + o_kbuf, dsw + (size_t)(s - 1) * 768 * 3072, dsb + (s - 1) * 768,
          nullptr, cls, M, 768, 3072);
      clp = cls;
    }
    ln_kernel<<<M, 256, 0, stream>>>(clp, nullptr, nfg, nfb, ws + o_feat);
    gemm_wmma_f32<false, false, false><<<dim3(12, M / 128), 256, 0, stream>>>(
        ws + o_feat, kw + (size_t)s * 589824, kb + s * 768, nullptr,
        ws + o_kbuf, M, 768, 768);
    gemm_wmma_f32<false, false, false><<<dim3(12, M / 128), 256, 0, stream>>>(
        ws + o_feat, vw + (size_t)s * 589824, vb + s * 768, nullptr,
        ws + o_vbuf, M, 768, 768);
    gemm_wmma_f32<false, false, false><<<dim3(12, 7), 256, 0, stream>>>(
        qtok, qw + (size_t)s * 589824, qb + s * 768, nullptr, ws + o_qbuf,
        800, 768, 768);
    const float sg = sigmas[s];
    bias_k<<<(8 * HW + 255) / 256, 256, 0, stream>>>(
        kpts, ws + o_bias, Hc, Hc, 1.0f / (2.0f * sg * sg), HW);
    attn_kernel<<<64, 256, 0, stream>>>(ws + o_qbuf, ws + o_kbuf, ws + o_vbuf,
                                        ws + o_bias, ws + o_attn, HW, qscale);
    gemm_wmma_f32<false, false, true><<<dim3(12, 7), 256, 0, stream>>>(
        ws + o_attn, ow + (size_t)s * 589824, ob + s * 768, ws + o_wsm + s,
        ws + o_fus, 800, 768, 768);
  }

  // x = LN(query + fused)
  ln_kernel<<<800, 256, 0, stream>>>(ws + o_fus, qtok, nag, nab, ws + o_x);
  // hid = relu(x @ W1 + b1)
  gemm_wmma_f32<false, true, false><<<dim3(48, 7), 256, 0, stream>>>(
      ws + o_x, fw1, fb1, nullptr, ws + o_hid, 800, 3072, 768);
  // y = x + hid @ W2 + b2
  hipMemcpyAsync(ws + o_y, ws + o_x, 800ull * 768 * sizeof(float),
                 hipMemcpyDeviceToDevice, stream);
  gemm_wmma_f32<false, false, true><<<dim3(12, 7), 256, 0, stream>>>(
      ws + o_hid, fw2, fb2, nullptr, ws + o_y, 800, 768, 3072);
  // out = LN(y)
  ln_kernel<<<800, 256, 0, stream>>>(ws + o_y, nullptr, ng2, nb2, out);
}
